// BayesLSTMModel_68083821576398
// MI455X (gfx1250) — matrix operand order repaired and checked by reading, not compiled, not run on hardware
//
#include <hip/hip_runtime.h>
#include <hip/hip_bf16.h>
#include <math.h>

// ---------------------------------------------------------------------------
// BayesLSTM forward for MI455X (gfx1250, wave32, WMMA).
// GEMMs: v_wmma_f32_16x16x32_bf16, fp32 operands converted via
// v_cvt_pk_bf16_f32 while staging to LDS, f32 accumulation.
// Software-pipelined: global loads (k+1) -> WMMA (k) -> LDS commit (k+1).
// Recurrence: one fused kernel per (layer,t): 8 waves = {W_hh,Wb_h} x 4 gates
// of 32x32 tiles over K=1024, + fused cell update.
// ---------------------------------------------------------------------------

typedef __attribute__((ext_vector_type(16))) __bf16 v16bf;
typedef __attribute__((ext_vector_type(8)))  float  v8f;

#define HID   1024
#define SB    4096     // S*B rows
#define BATCH 32
#define SEQ   128
#define VOC   32000
#define LROW  17       // LDS dwords per 32-element bf16 row (16 + 1 pad)

__device__ __forceinline__ unsigned cvt_pk(float a, float b) {
  unsigned r;
  asm("v_cvt_pk_bf16_f32 %0, %1, %2" : "=v"(r) : "v"(a), "v"(b));
  return r;
}

union Frag { unsigned u[8]; v16bf v; };

// ---------------------------------------------------------------------------
// Big GEMM: C[M,N] = A[M,K] * B[N,K]^T (+bias[N]); block tile 128x128,
// 8 waves, wave tile 32x64 (2x4 WMMA accumulators).
// Thread t stages A(row t>>1, half t&1) and B(row t>>1, half t&1): static.
// ---------------------------------------------------------------------------
__global__ __launch_bounds__(256)
void gemm128(const float* __restrict__ A, int lda,
             const float* __restrict__ B, int ldb,
             float* __restrict__ C, long long ldc,
             const float* __restrict__ bias,
             int M, int N, int K) {
  __shared__ unsigned lds[2 * 256 * LROW];        // 2 bufs x (128 A + 128 B)

  const int tid  = threadIdx.x;
  const int wave = tid >> 5, lane = tid & 31;
  const int wm = wave & 3, wn = wave >> 2;        // 4x2 wave grid
  const int lo = lane & 15, hi = lane >> 4;
  const int m0 = blockIdx.y * 128;
  const int n0 = blockIdx.x * 128;

  const int rowi = tid >> 1, half = tid & 1;
  int ga = m0 + rowi; if (ga > M - 1) ga = M - 1;
  int gb = n0 + rowi; if (gb > N - 1) gb = N - 1;
  const float* spA = A + (size_t)ga * lda + half * 16;
  const float* spB = B + (size_t)gb * ldb + half * 16;
  const int dA = rowi * LROW + half * 8;
  const int dB = (128 + rowi) * LROW + half * 8;

  v8f acc[2][4];
#pragma unroll
  for (int i = 0; i < 2; ++i)
#pragma unroll
    for (int j = 0; j < 4; ++j)
#pragma unroll
      for (int v = 0; v < 8; ++v) acc[i][j][v] = 0.f;

  float4 ra[4], rb[4];
  auto ldregs = [&]() {
#pragma unroll
    for (int i = 0; i < 4; ++i) ra[i] = ((const float4*)spA)[i];
#pragma unroll
    for (int i = 0; i < 4; ++i) rb[i] = ((const float4*)spB)[i];
    spA += 32; spB += 32;
  };
  auto commit = [&](unsigned* buf) {
#pragma unroll
    for (int i = 0; i < 4; ++i) {
      buf[dA + 2 * i]     = cvt_pk(ra[i].x, ra[i].y);
      buf[dA + 2 * i + 1] = cvt_pk(ra[i].z, ra[i].w);
      buf[dB + 2 * i]     = cvt_pk(rb[i].x, rb[i].y);
      buf[dB + 2 * i + 1] = cvt_pk(rb[i].z, rb[i].w);
    }
  };
  auto compute = [&](const unsigned* buf) {
    Frag a[2], b[4];
#pragma unroll
    for (int i = 0; i < 2; ++i) {
      const unsigned* p = buf + (wm * 32 + i * 16 + lo) * LROW + hi * 4;
#pragma unroll
      for (int v = 0; v < 8; ++v) a[i].u[v] = p[(v & 3) + ((v >> 2) << 3)];
    }
#pragma unroll
    for (int j = 0; j < 4; ++j) {
      const unsigned* p = buf + (128 + wn * 64 + j * 16 + lo) * LROW + hi * 4;
#pragma unroll
      for (int v = 0; v < 8; ++v) b[j].u[v] = p[(v & 3) + ((v >> 2) << 3)];
    }
#pragma unroll
    for (int i = 0; i < 2; ++i)
#pragma unroll
      for (int j = 0; j < 4; ++j)
        acc[i][j] = __builtin_amdgcn_wmma_f32_16x16x32_bf16(
            false, a[i].v, false, b[j].v, (short)0, acc[i][j], false, false);
  };

  unsigned* buf0 = lds;
  unsigned* buf1 = lds + 256 * LROW;
  ldregs();
  commit(buf0);
  __syncthreads();
  const int nk = K >> 5;
  for (int kk = 0; kk < nk; ++kk) {
    unsigned* cur = (kk & 1) ? buf1 : buf0;
    unsigned* nxt = (kk & 1) ? buf0 : buf1;
    const bool more = (kk + 1 < nk);
    if (more) ldregs();          // issue global loads for next slab
    compute(cur);                // WMMA on current slab (hides load latency)
    if (more) commit(nxt);       // convert + LDS store
    __syncthreads();
  }

#pragma unroll
  for (int i = 0; i < 2; ++i) {
    const int rbase = m0 + wm * 32 + i * 16 + hi * 8;
#pragma unroll
    for (int j = 0; j < 4; ++j) {
      const int col = n0 + wn * 64 + j * 16 + lo;
      if (col < N) {
        const float bv = bias ? bias[col] : 0.f;
#pragma unroll
        for (int v = 0; v < 8; ++v) {
          const int r = rbase + v;
          if (r < M) C[(size_t)r * ldc + col] = acc[i][j][v] + bv;
        }
      }
    }
  }
}

// ---------------------------------------------------------------------------
// Fused recurrent step: grid.x = 32 blocks (32 hidden units each), 256 thr.
// Wave w -> matrix (w>>2: 0=W_hh, 1=Wb_h) x gate (w&3): 32x32 tile, K=1024.
// Staging: each lane stages one full 32-float W row; lanes 0-3 per wave also
// stage one h row. Then fused cell update for the block's 32x32 elements.
// ---------------------------------------------------------------------------
__global__ __launch_bounds__(256)
void lstm_step_k(const float* __restrict__ hprev, // [B,H]
                 const float* __restrict__ Whh,   // [4H,H]
                 const float* __restrict__ Wbh,   // rows [4H], ld 2048
                 const float* __restrict__ Xg_t,  // [B,4H]
                 const float* __restrict__ Xb_t,  // [B,4H]
                 const float* __restrict__ bih,   // [4H]
                 const float* __restrict__ bb,    // [4H]
                 const float* __restrict__ mix,   // [4,2]
                 float* __restrict__ c,           // [B,H] in-place
                 float* __restrict__ hout) {      // [B,H] (= o_l row t)
  constexpr int ROWS = 32 + 256;                  // 32 h rows + 8*32 W rows
  __shared__ unsigned lds[2 * ROWS * LROW];       // 39168 B; reused as cellbuf

  const int tid  = threadIdx.x;
  const int wave = tid >> 5, lane = tid & 31;
  const int lo = lane & 15, hi = lane >> 4;
  const int j0 = blockIdx.x * 32;

  // B row for this lane (full 32-float row per K slab)
  const size_t wr = (size_t)((wave & 3) * HID + j0 + lane);
  const float* spB = (wave < 4) ? (Whh + wr * HID) : (Wbh + wr * 2048);
  const int dB = (32 + wave * 32 + lane) * LROW;
  // A (h) row for lanes 0..3 of each wave
  const bool hasA = (lane < 4);
  const int rowA = wave * 4 + lane;
  const float* spA = hprev + (size_t)rowA * HID;
  const int dA = rowA * LROW;

  v8f acc[2][2];
#pragma unroll
  for (int i = 0; i < 2; ++i)
#pragma unroll
    for (int j = 0; j < 2; ++j)
#pragma unroll
      for (int v = 0; v < 8; ++v) acc[i][j][v] = 0.f;

  float4 rb[8], ra[8];
  auto ldregs = [&]() {
#pragma unroll
    for (int i = 0; i < 8; ++i) rb[i] = ((const float4*)spB)[i];
    if (hasA) {
#pragma unroll
      for (int i = 0; i < 8; ++i) ra[i] = ((const float4*)spA)[i];
    }
    spB += 32; spA += 32;
  };
  auto commit = [&](unsigned* buf) {
#pragma unroll
    for (int i = 0; i < 8; ++i) {
      buf[dB + 2 * i]     = cvt_pk(rb[i].x, rb[i].y);
      buf[dB + 2 * i + 1] = cvt_pk(rb[i].z, rb[i].w);
    }
    if (hasA) {
#pragma unroll
      for (int i = 0; i < 8; ++i) {
        buf[dA + 2 * i]     = cvt_pk(ra[i].x, ra[i].y);
        buf[dA + 2 * i + 1] = cvt_pk(ra[i].z, ra[i].w);
      }
    }
  };
  auto compute = [&](const unsigned* buf) {
    Frag a[2], b[2];
#pragma unroll
    for (int i = 0; i < 2; ++i) {
      const unsigned* p = buf + (i * 16 + lo) * LROW + hi * 4;
#pragma unroll
      for (int v = 0; v < 8; ++v) a[i].u[v] = p[(v & 3) + ((v >> 2) << 3)];
    }
#pragma unroll
    for (int j = 0; j < 2; ++j) {
      const unsigned* p = buf + (32 + wave * 32 + j * 16 + lo) * LROW + hi * 4;
#pragma unroll
      for (int v = 0; v < 8; ++v) b[j].u[v] = p[(v & 3) + ((v >> 2) << 3)];
    }
#pragma unroll
    for (int i = 0; i < 2; ++i)
#pragma unroll
      for (int j = 0; j < 2; ++j)
        acc[i][j] = __builtin_amdgcn_wmma_f32_16x16x32_bf16(
            false, a[i].v, false, b[j].v, (short)0, acc[i][j], false, false);
  };

  unsigned* buf0 = lds;
  unsigned* buf1 = lds + ROWS * LROW;
  ldregs();
  commit(buf0);
  __syncthreads();
  for (int kk = 0; kk < 32; ++kk) {
    unsigned* cur = (kk & 1) ? buf1 : buf0;
    unsigned* nxt = (kk & 1) ? buf0 : buf1;
    const bool more = (kk + 1 < 32);
    if (more) ldregs();
    compute(cur);
    if (more) commit(nxt);
    __syncthreads();
  }

  // dump accumulators: cb[w*1024 + b*32 + jj]  (conflict-free for update)
  float* cb = (float*)lds;
#pragma unroll
  for (int i = 0; i < 2; ++i) {
#pragma unroll
    for (int j = 0; j < 2; ++j) {
      const int jj = j * 16 + lo;
      const int bb0 = i * 16 + hi * 8;
#pragma unroll
      for (int v = 0; v < 8; ++v)
        cb[wave * 1024 + (bb0 + v) * 32 + jj] = acc[i][j][v];
    }
  }
  __syncthreads();

  // fused cell update: 32 units x 32 batch = 1024 elements
  for (int e = tid; e < 1024; e += 256) {
    const int jj = e & 31, b = e >> 5;
    const int col = j0 + jj;
    float pre[4], bay[4], p0[4], p1[4];
#pragma unroll
    for (int g = 0; g < 4; ++g) {
      pre[g] = cb[g * 1024 + b * 32 + jj]
             + Xg_t[(size_t)b * 4096 + g * HID + col] + 2.f * bih[g * HID + col];
      bay[g] = cb[(4 + g) * 1024 + b * 32 + jj]
             + Xb_t[(size_t)b * 4096 + g * HID + col] + bb[g * HID + col];
      const float e0 = __expf(mix[2 * g]), e1 = __expf(mix[2 * g + 1]);
      const float inv = 1.f / (e0 + e1);
      p0[g] = e0 * inv; p1[g] = e1 * inv;
    }
    auto sig = [](float v) { return 1.f / (1.f + __expf(-v)); };
    const float ig = sig(pre[0]) * p0[0] + sig(bay[0]) * p1[0];
    const float fg = sig(pre[1]) * p0[1] + sig(bay[1]) * p1[1];
    const float gg = tanhf(pre[2]) * p0[2] + tanhf(bay[2]) * p1[2];
    const float og = sig(pre[3]) * p0[3] + sig(bay[3]) * p1[3];
    const int ci = b * HID + col;
    const float cn = fg * c[ci] + ig * gg;
    c[ci] = cn;
    hout[ci] = og * tanhf(cn);
  }
}

// Embedding gather: out[row, col] = emb_W[x[row], col]
__global__ void embed_k(const int* __restrict__ x,
                        const float* __restrict__ embW,
                        float* __restrict__ out) {
  const int idx = blockIdx.x * blockDim.x + threadIdx.x;
  const int row = idx >> 10, col = idx & 1023;
  out[idx] = embW[(size_t)x[row] * HID + col];
}

__global__ void copy2_k(const float* __restrict__ a, const float* __restrict__ b,
                        float* __restrict__ oa, float* __restrict__ ob, int n) {
  const int i = blockIdx.x * blockDim.x + threadIdx.x;
  if (i < n) { oa[i] = a[i]; ob[i] = b[i]; }
}

extern "C" void kernel_launch(void* const* d_in, const int* in_sizes, int n_in,
                              void* d_out, int out_size, void* d_ws, size_t ws_size,
                              hipStream_t stream) {
  const int*   x    = (const int*)  d_in[0];
  const float* h0   = (const float*)d_in[1];
  const float* c0   = (const float*)d_in[2];
  const float* embW = (const float*)d_in[3];
  const float* decW = (const float*)d_in[4];
  const float* decb = (const float*)d_in[5];
  const float* W_ih = (const float*)d_in[6];
  const float* b_ih = (const float*)d_in[7];
  const float* W_hh = (const float*)d_in[8];
  const float* Wb   = (const float*)d_in[9];
  const float* bb   = (const float*)d_in[10];
  const float* mix  = (const float*)d_in[11];
  float* out = (float*)d_out;

  // workspace layout (fp32 elements)
  float* ws  = (float*)d_ws;
  float* emb = ws;                                  // SB*H
  float* o0  = emb + (size_t)SB * HID;              // SB*H  (layer-0 h per t)
  float* o1  = o0  + (size_t)SB * HID;              // SB*H  (layer-1 h per t)
  float* Xg  = o1  + (size_t)SB * HID;              // SB*4H
  float* Xb  = Xg  + (size_t)SB * 4096;             // SB*4H
  float* cs  = Xb  + (size_t)SB * 4096;             // 2*B*H (cell state)

  float* decoded = out;                             // [SB, V]
  float* out_h   = out + (size_t)SB * VOC;          // [2,B,H]
  float* out_c   = out_h + 2 * BATCH * HID;         // [2,B,H]

  embed_k<<<(SB * HID) / 256, 256, 0, stream>>>(x, embW, emb);
  copy2_k<<<(2 * BATCH * HID) / 256, 256, 0, stream>>>(
      c0, c0 + BATCH * HID, cs, cs + BATCH * HID, BATCH * HID);

  const float* layer_in[2]  = { emb, o0 };
  float*       layer_out[2] = { o0, o1 };

  for (int l = 0; l < 2; ++l) {
    const float* Wih_l = W_ih + (size_t)l * 4096 * HID;
    const float* Whh_l = W_hh + (size_t)l * 4096 * HID;
    const float* Wb_l  = Wb   + (size_t)l * 4096 * 2048;  // [4H rows, ld 2048]
    const float* bih_l = b_ih + l * 4096;
    const float* bb_l  = bb   + l * 4096;
    const float* mix_l = mix  + l * 8;
    const float* Ain = layer_in[l];
    float* Ol = layer_out[l];
    float* cl = cs + l * BATCH * HID;

    // precompute input-dependent parts for all timesteps (big GEMMs)
    gemm128<<<dim3(4096 / 128, SB / 128), 256, 0, stream>>>(
        Ain, HID, Wih_l, HID, Xg, 4096, nullptr, SB, 4096, HID);
    gemm128<<<dim3(4096 / 128, SB / 128), 256, 0, stream>>>(
        Ain, HID, Wb_l, 2048, Xb, 4096, nullptr, SB, 4096, HID);

    for (int t = 0; t < SEQ; ++t) {
      const float* hp = (t == 0) ? (h0 + (size_t)l * BATCH * HID)
                                 : (Ol + (size_t)(t - 1) * BATCH * HID);
      lstm_step_k<<<32, 256, 0, stream>>>(
          hp, Whh_l, Wb_l + HID,
          Xg + (size_t)t * BATCH * 4096, Xb + (size_t)t * BATCH * 4096,
          bih_l, bb_l, mix_l, cl, Ol + (size_t)t * BATCH * HID);
    }
  }

  // decode: o1 @ dec_W^T + dec_b
  gemm128<<<dim3((VOC + 127) / 128, SB / 128), 256, 0, stream>>>(
      o1, HID, decW, HID, decoded, VOC, decb, SB, VOC, HID);

  // final states: hids from last-step h rows, cells from cs
  copy2_k<<<(BATCH * HID) / 256, 256, 0, stream>>>(
      o0 + (size_t)(SEQ - 1) * BATCH * HID, o1 + (size_t)(SEQ - 1) * BATCH * HID,
      out_h, out_h + BATCH * HID, BATCH * HID);
  copy2_k<<<(BATCH * HID) / 256, 256, 0, stream>>>(
      cs, cs + BATCH * HID, out_c, out_c + BATCH * HID, BATCH * HID);
}
